// MyLSTM_50981261803694
// MI455X (gfx1250) — matrix operand order repaired and checked by reading, not compiled, not run on hardware
//
#include <hip/hip_runtime.h>
#include <hip/hip_bf16.h>

// ---------------- problem constants ----------------
static constexpr int BB  = 64;      // batch
static constexpr int TT  = 196;     // timesteps
static constexpr int IN  = 512;     // input dim
static constexpr int HH  = 512;     // hidden dim
static constexpr int GG  = 2048;    // 4*H gates
static constexpr int VV  = 5000;    // vocab
static constexpr int VP  = 5056;    // vocab padded to 64*79

typedef __attribute__((ext_vector_type(16))) __bf16 v16bf;
typedef __attribute__((ext_vector_type(8)))  float  v8f;

// ---------------- WMMA fragment loaders (contiguous k = 16*half + e in both A and B) ----------------
// A (16x32 bf16, row-major src, stride lda): lane l -> m = l&15, half = l>>4
//   elems 0..7  : k = k0 + half*16 + e      (16 contiguous bytes)
//   elems 8..15 : k = k0 + half*16 + 8 + e  (16 contiguous bytes)
__device__ __forceinline__ v16bf load_frag_a(const __bf16* base, int lda,
                                             int m, int k0, int half) {
  union { v16bf v; uint4 q[2]; } f;
  const __bf16* p0 = base + (size_t)m * lda + k0 + half * 16;
  const __bf16* p1 = base + (size_t)m * lda + k0 + half * 16 + 8;
  f.q[0] = *(const uint4*)p0;
  f.q[1] = *(const uint4*)p1;
  return f.v;
}

// B (32x16 bf16) built from W (N x K row-major): output col n <- W row n.
// lane l -> n = l&15, half = l>>4 ; elem e -> k = k0 + half*16 + e
// => 32 contiguous bf16 from one W row.
__device__ __forceinline__ v16bf load_frag_b(const __bf16* wrow, int k0, int half) {
  union { v16bf v; uint4 q[2]; } f;
  const uint4* p = (const uint4*)(wrow + k0 + half * 16);
  f.q[0] = p[0];
  f.q[1] = p[1];
  return f.v;
}

__device__ __forceinline__ v8f zero_v8f() {
  v8f z;
#pragma unroll
  for (int i = 0; i < 8; ++i) z[i] = 0.0f;
  return z;
}

__device__ __forceinline__ float sigmoidf_(float x) {
  return 1.0f / (1.0f + __expf(-x));
}

// ---------------- small prep kernels ----------------
__global__ void cvt_f32_to_bf16_pad(const float* __restrict__ s,
                                    __bf16* __restrict__ d, int nsrc, int ndst) {
  int i = blockIdx.x * 256 + threadIdx.x;
  if (i < ndst) d[i] = (i < nsrc) ? (__bf16)s[i] : (__bf16)0.0f;
}

__global__ void bias_combine(const float* __restrict__ a,
                             const float* __restrict__ b,
                             float* __restrict__ d, int n) {
  int i = blockIdx.x * 256 + threadIdx.x;
  if (i < n) d[i] = a[i] + b[i];
}

__global__ void bias_pad(const float* __restrict__ a, float* __restrict__ d,
                         int nsrc, int ndst) {
  int i = blockIdx.x * 256 + threadIdx.x;
  if (i < ndst) d[i] = (i < nsrc) ? a[i] : 0.0f;
}

// x: (B, IN, 14*14) f32  ->  xseq: rows r = t*B + b, cols i  (bf16)
__global__ void transpose_x(const float* __restrict__ x, __bf16* __restrict__ xseq) {
  int idx = blockIdx.x * 256 + threadIdx.x;
  if (idx >= BB * TT * IN) return;
  int i = idx % IN;
  int r = idx / IN;
  int b = r % BB;
  int t = r / BB;
  xseq[idx] = (__bf16)x[((size_t)b * IN + i) * TT + t];
}

// ---------------- big GEMM:  out = A(MxK) @ W(NxK)^T + bias ----------------
// MODE 0: out[m*ldn + n]                (input projections, f32)
// MODE 1: out[(b*T + t)*V + n], m=t*B+b, guarded n<VV  (classifier)
template <int MODE>
__global__ void gemm_bias(const __bf16* __restrict__ A, const __bf16* __restrict__ W,
                          const float* __restrict__ bias, float* __restrict__ out,
                          int K, int ldn) {
  const int lane = threadIdx.x & 31;
  const int wave = threadIdx.x >> 5;         // 4 waves
  const int half = lane >> 4;
  const int nl   = lane & 15;

  const int mRow  = blockIdx.y * 64 + wave * 16;  // 16 rows per wave
  const int nBase = blockIdx.x * 64;              // 64 cols per WG

  v8f acc[4];
#pragma unroll
  for (int ni = 0; ni < 4; ++ni) acc[ni] = zero_v8f();

  for (int k0 = 0; k0 < K; k0 += 32) {
    v16bf a = load_frag_a(A, K, mRow + nl, k0, half);
#pragma unroll
    for (int ni = 0; ni < 4; ++ni) {
      const __bf16* wrow = W + (size_t)(nBase + ni * 16 + nl) * K;
      v16bf b = load_frag_b(wrow, k0, half);
      acc[ni] = __builtin_amdgcn_wmma_f32_16x16x32_bf16(
          false, a, false, b, (short)0, acc[ni], false, false);
    }
  }

#pragma unroll
  for (int ni = 0; ni < 4; ++ni) {
    int n = nBase + ni * 16 + nl;
    float bv = bias[n];
#pragma unroll
    for (int j = 0; j < 8; ++j) {
      int m = mRow + j + 8 * half;
      float v = acc[ni][j] + bv;
      if (MODE == 0) {
        out[(size_t)m * ldn + n] = v;
      } else {
        if (n < VV) {
          int t = m / BB, b = m % BB;
          out[((size_t)b * TT + t) * VV + n] = v;
        }
      }
    }
  }
}

// ---------------- LSTM timestep ----------------
// grid: 16 WGs (32 hidden units each), block: 128 (4 waves = 4 gates)
// wave w computes gate-w 64x32 tile:  g = xp_t + h_prev @ Whh^T
__global__ void lstm_step(const float* __restrict__ xp_t,      // (64, 2048) f32
                          const __bf16* __restrict__ Whh,      // (2048, 512) bf16
                          const __bf16* __restrict__ h_prev,   // (64, 512) bf16
                          float* __restrict__ c,               // (64, 512) f32
                          __bf16* __restrict__ h_out) {        // (64, 512) bf16
  __shared__ float sg[4][64][32];   // 32 KB

  const int lane = threadIdx.x & 31;
  const int wave = threadIdx.x >> 5;   // gate index: 0=i 1=f 2=g 3=o
  const int half = lane >> 4;
  const int nl   = lane & 15;
  const int n0   = blockIdx.x * 32;    // hidden chunk base

  v8f acc[4][2];
#pragma unroll
  for (int mi = 0; mi < 4; ++mi)
#pragma unroll
    for (int ni = 0; ni < 2; ++ni) acc[mi][ni] = zero_v8f();

  for (int k0 = 0; k0 < HH; k0 += 32) {
    v16bf bfrag[2];
#pragma unroll
    for (int ni = 0; ni < 2; ++ni) {
      const __bf16* wrow = Whh + (size_t)(wave * HH + n0 + ni * 16 + nl) * HH;
      bfrag[ni] = load_frag_b(wrow, k0, half);
    }
#pragma unroll
    for (int mi = 0; mi < 4; ++mi) {
      v16bf a = load_frag_a(h_prev, HH, mi * 16 + nl, k0, half);
#pragma unroll
      for (int ni = 0; ni < 2; ++ni) {
        acc[mi][ni] = __builtin_amdgcn_wmma_f32_16x16x32_bf16(
            false, a, false, bfrag[ni], (short)0, acc[mi][ni], false, false);
      }
    }
  }

  // add precomputed input projection, park gates in LDS
#pragma unroll
  for (int mi = 0; mi < 4; ++mi) {
#pragma unroll
    for (int ni = 0; ni < 2; ++ni) {
      int nlocal = ni * 16 + nl;
#pragma unroll
      for (int j = 0; j < 8; ++j) {
        int m = mi * 16 + j + 8 * half;           // batch index 0..63
        float v = acc[mi][ni][j] +
                  xp_t[(size_t)m * GG + wave * HH + n0 + nlocal];
        sg[wave][m][nlocal] = v;
      }
    }
  }
  __syncthreads();

  // cell update: 64x32 elements, 128 threads -> 16 each
  for (int e = threadIdx.x; e < 64 * 32; e += 128) {
    int b  = e >> 5;
    int nn = e & 31;
    int ng = n0 + nn;
    float gi = sigmoidf_(sg[0][b][nn]);
    float gf = sigmoidf_(sg[1][b][nn]);
    float gg = tanhf(sg[2][b][nn]);
    float go = sigmoidf_(sg[3][b][nn]);
    float cn = gf * c[(size_t)b * HH + ng] + gi * gg;
    c[(size_t)b * HH + ng] = cn;
    h_out[(size_t)b * HH + ng] = (__bf16)(go * tanhf(cn));
  }
}

// ---------------- host ----------------
extern "C" void kernel_launch(void* const* d_in, const int* in_sizes, int n_in,
                              void* d_out, int out_size, void* d_ws, size_t ws_size,
                              hipStream_t stream) {
  (void)in_sizes; (void)n_in; (void)out_size; (void)ws_size;

  const float* x     = (const float*)d_in[0];
  const float* Wih0  = (const float*)d_in[1];
  const float* Whh0  = (const float*)d_in[2];
  const float* bih0  = (const float*)d_in[3];
  const float* bhh0  = (const float*)d_in[4];
  const float* Wih1  = (const float*)d_in[5];
  const float* Whh1  = (const float*)d_in[6];
  const float* bih1  = (const float*)d_in[7];
  const float* bhh1  = (const float*)d_in[8];
  const float* Wfc   = (const float*)d_in[9];
  const float* bfc   = (const float*)d_in[10];
  float* out = (float*)d_out;

  // workspace carve-up (256B aligned)
  char* ws = (char*)d_ws;
  size_t off = 0;
  auto carve = [&](size_t bytes) {
    char* p = ws + off;
    off += (bytes + 255) & ~(size_t)255;
    return p;
  };
  __bf16* xseq  = (__bf16*)carve((size_t)TT * BB * IN * 2);
  __bf16* hall0 = (__bf16*)carve((size_t)TT * BB * HH * 2);
  __bf16* hall1 = (__bf16*)carve((size_t)TT * BB * HH * 2);
  __bf16* wih0b = (__bf16*)carve((size_t)GG * IN * 2);
  __bf16* whh0b = (__bf16*)carve((size_t)GG * HH * 2);
  __bf16* wih1b = (__bf16*)carve((size_t)GG * HH * 2);
  __bf16* whh1b = (__bf16*)carve((size_t)GG * HH * 2);
  __bf16* wfcb  = (__bf16*)carve((size_t)VP * HH * 2);
  float*  bias0 = (float*)carve((size_t)GG * 4);
  float*  bias1 = (float*)carve((size_t)GG * 4);
  float*  bfcp  = (float*)carve((size_t)VP * 4);
  float*  xp    = (float*)carve((size_t)TT * BB * GG * 4);
  float*  cbuf  = (float*)carve((size_t)BB * HH * 4);
  __bf16* h0z   = (__bf16*)carve((size_t)BB * HH * 2);

  auto nblk = [](size_t n) { return (unsigned)((n + 255) / 256); };

  // 1) weight/bias prep
  cvt_f32_to_bf16_pad<<<nblk((size_t)GG * IN), 256, 0, stream>>>(Wih0, wih0b, GG * IN, GG * IN);
  cvt_f32_to_bf16_pad<<<nblk((size_t)GG * HH), 256, 0, stream>>>(Whh0, whh0b, GG * HH, GG * HH);
  cvt_f32_to_bf16_pad<<<nblk((size_t)GG * HH), 256, 0, stream>>>(Wih1, wih1b, GG * HH, GG * HH);
  cvt_f32_to_bf16_pad<<<nblk((size_t)GG * HH), 256, 0, stream>>>(Whh1, whh1b, GG * HH, GG * HH);
  cvt_f32_to_bf16_pad<<<nblk((size_t)VP * HH), 256, 0, stream>>>(Wfc, wfcb, VV * HH, VP * HH);
  bias_combine<<<nblk(GG), 256, 0, stream>>>(bih0, bhh0, bias0, GG);
  bias_combine<<<nblk(GG), 256, 0, stream>>>(bih1, bhh1, bias1, GG);
  bias_pad<<<nblk(VP), 256, 0, stream>>>(bfc, bfcp, VV, VP);
  hipMemsetAsync(h0z, 0, (size_t)BB * HH * 2, stream);

  // 2) x -> (T*B, IN) bf16
  transpose_x<<<nblk((size_t)BB * TT * IN), 256, 0, stream>>>(x, xseq);

  dim3 blkG(128);
  dim3 grdXP(GG / 64, (TT * BB) / 64);   // 32 x 196

  // ---- layer 0 ----
  gemm_bias<0><<<grdXP, blkG, 0, stream>>>(xseq, wih0b, bias0, xp, IN, GG);
  hipMemsetAsync(cbuf, 0, (size_t)BB * HH * 4, stream);
  for (int t = 0; t < TT; ++t) {
    const __bf16* hprev = (t == 0) ? h0z : (hall0 + (size_t)(t - 1) * BB * HH);
    lstm_step<<<16, 128, 0, stream>>>(xp + (size_t)t * BB * GG, whh0b, hprev,
                                      cbuf, hall0 + (size_t)t * BB * HH);
  }

  // ---- layer 1 ----
  gemm_bias<0><<<grdXP, blkG, 0, stream>>>(hall0, wih1b, bias1, xp, HH, GG);
  hipMemsetAsync(cbuf, 0, (size_t)BB * HH * 4, stream);
  for (int t = 0; t < TT; ++t) {
    const __bf16* hprev = (t == 0) ? h0z : (hall1 + (size_t)(t - 1) * BB * HH);
    lstm_step<<<16, 128, 0, stream>>>(xp + (size_t)t * BB * GG, whh1b, hprev,
                                      cbuf, hall1 + (size_t)t * BB * HH);
  }

  // ---- classifier ----
  dim3 grdFC(VP / 64, (TT * BB) / 64);   // 79 x 196
  gemm_bias<1><<<grdFC, blkG, 0, stream>>>(hall1, wfcb, bfcp, out, HH, VV);
}
